// Self_Attn_18631568130694
// MI455X (gfx1250) — matrix operand order repaired and checked by reading, not compiled, not run on hardware
//
#include <hip/hip_runtime.h>

// ---------- types ----------
typedef __attribute__((ext_vector_type(16))) __bf16 v16bf;
typedef __attribute__((ext_vector_type(8)))  float  v8f;

union BF16x16 {
    v16bf v;
    uint4 q[2];
    unsigned short u[16];
};

__device__ __forceinline__ unsigned short f2bf(float f) {
    unsigned int u = __float_as_uint(f);
    unsigned int r = (u + 0x7FFFu + ((u >> 16) & 1u)) >> 16;
    return (unsigned short)r;
}
__device__ __forceinline__ float bf2f(unsigned short u) {
    return __uint_as_float(((unsigned int)u) << 16);
}

__device__ __forceinline__ v8f wmma_bf16(const BF16x16& a, const BF16x16& b, v8f c) {
    return __builtin_amdgcn_wmma_f32_16x16x32_bf16(false, a.v, false, b.v,
                                                   (short)0, c, false, false);
}

// ---------- problem constants ----------
#define BB   16
#define CC   256
#define NN   4096   // 64*64
#define C8   32
#define C2   128
#define MM   1024   // pooled pixels

// ---------- spectral norm: sigma = ||W (W^T u / ||W^T u||)||, out = bf16(W/sigma) ----------
__device__ __forceinline__ float block_sum256(float v, float* red) {
    int t = threadIdx.x;
    red[t] = v;
    __syncthreads();
    for (int s = 128; s > 0; s >>= 1) {
        if (t < s) red[t] += red[t + s];
        __syncthreads();
    }
    float r = red[0];
    __syncthreads();
    return r;
}

__global__ void sn_scale_kernel(const float* __restrict__ w, const float* __restrict__ u,
                                unsigned short* __restrict__ out, int O, int I) {
    __shared__ float red[256];
    __shared__ float sv[256];
    int t = threadIdx.x;
    float vi = 0.f;
    if (t < I) {
        for (int o = 0; o < O; ++o) vi += w[o * I + t] * u[o];
    }
    float n1 = block_sum256(vi * vi, red);
    vi *= rsqrtf(fmaxf(n1, 1e-30f));
    sv[t] = (t < I) ? vi : 0.f;
    __syncthreads();
    float to = 0.f;
    if (t < O) {
        for (int i = 0; i < I; ++i) to += w[t * I + i] * sv[i];
    }
    float n2 = block_sum256((t < O) ? to * to : 0.f, red);
    float inv = rsqrtf(fmaxf(n2, 1e-30f));   // 1/sigma
    for (int e = t; e < O * I; e += 256) out[e] = f2bf(w[e] * inv);
}

// ---------- convert + transpose: x [B][C][N] f32  ->  xt [B][N][C] bf16 ----------
// One block per (b, 16-channel tile, 64-pixel tile). Coalesced f32 reads,
// LDS transpose, vectorized 8-byte bf16 writes.
__global__ void cvt_transpose_kernel(const float* __restrict__ x,
                                     unsigned short* __restrict__ xt) {
    __shared__ unsigned short tile[64][20];   // [n_local][c_local], padded
    int bt = blockIdx.x;
    int nt = bt & 63;
    int ct = (bt >> 6) & 15;
    int b  = bt >> 10;
    int n0 = nt * 64;
    int c0 = ct * 16;
    int t  = threadIdx.x;

    int nl = t & 63;
    int cb = t >> 6;          // 0..3
#pragma unroll
    for (int it = 0; it < 4; ++it) {
        int cl = cb + it * 4;
        tile[nl][cl] = f2bf(x[((size_t)(b * CC + c0 + cl)) * NN + n0 + nl]);
    }
    __syncthreads();

    int n  = t >> 2;          // 0..63
    int cp = (t & 3) * 4;     // 0,4,8,12
    uint2 pk;
    pk.x = (unsigned int)tile[n][cp + 0] | ((unsigned int)tile[n][cp + 1] << 16);
    pk.y = (unsigned int)tile[n][cp + 2] | ((unsigned int)tile[n][cp + 3] << 16);
    *(uint2*)(xt + ((size_t)(b * NN + n0 + n)) * CC + c0 + cp) = pk;
}

// ---------- generic 1x1 conv as WMMA GEMM ----------
// y[b,n,o] = sum_i Ws[o,i] * xt[b,n,i] + bias[o]   (all bf16 in, y bf16 [B][N][O])
// 256 threads = 8 waves, one 16(n) x 16(o) tile per wave, K=256 fully unrolled (8 WMMA).
__global__ void conv1x1_kernel(const unsigned short* __restrict__ xt,
                               const unsigned short* __restrict__ wS,
                               const float* __restrict__ bias,
                               unsigned short* __restrict__ out, int O) {
    int wid  = blockIdx.x * 8 + (threadIdx.x >> 5);
    int lane = threadIdx.x & 31;
    int r    = lane & 15;
    int half = lane >> 4;
    int otiles = O >> 4;
    int ot = wid % otiles;
    int nt = (wid / otiles) & 255;
    int b  = wid / (otiles * 256);
    int n0 = nt * 16;
    int c0 = ot * 16;

    const unsigned short* xb = xt + (size_t)(b * NN + n0 + r) * CC;
    const unsigned short* wb = wS + (size_t)(c0 + r) * CC + half * 16;
    __builtin_prefetch(wb, 0, 0);

    v8f acc = {};
#pragma unroll
    for (int k0 = 0; k0 < CC; k0 += 32) {
        BF16x16 a;
        a.q[0] = *(const uint4*)(xb + k0 + half * 16);
        a.q[1] = *(const uint4*)(xb + k0 + half * 16 + 8);
        BF16x16 bm;
        bm.q[0] = *(const uint4*)(wb + k0);
        bm.q[1] = *(const uint4*)(wb + k0 + 8);
        acc = wmma_bf16(a, bm, acc);
    }
    float bv = bias[c0 + r];
#pragma unroll
    for (int v = 0; v < 8; ++v) {
        int M = v + half * 8;
        out[(size_t)(b * NN + n0 + M) * O + c0 + r] = f2bf(acc[v] + bv);
    }
}

// ---------- 2x2 maxpool: phi_full [B][N][32] -> phi_p [B][M][32] ----------
__global__ void pool_phi_kernel(const unsigned short* __restrict__ pf,
                                unsigned short* __restrict__ pp) {
    int idx = blockIdx.x * 256 + threadIdx.x;
    if (idx >= BB * MM * C8) return;
    int c = idx & 31;
    int m = (idx >> 5) & (MM - 1);
    int b = idx >> 15;
    int h2 = m >> 5, w2 = m & 31;
    int nb = (h2 * 2) * 64 + w2 * 2;
    size_t base = (size_t)b * NN * C8 + c;
    float a0 = bf2f(pf[base + (size_t)(nb)      * C8]);
    float a1 = bf2f(pf[base + (size_t)(nb + 1)  * C8]);
    float a2 = bf2f(pf[base + (size_t)(nb + 64) * C8]);
    float a3 = bf2f(pf[base + (size_t)(nb + 65) * C8]);
    pp[(size_t)(b * MM + m) * C8 + c] = f2bf(fmaxf(fmaxf(a0, a1), fmaxf(a2, a3)));
}

// ---------- 2x2 maxpool + transpose: g_full [B][N][128] -> g_p [B][128][M] ----------
__global__ void pool_g_kernel(const unsigned short* __restrict__ gf,
                              unsigned short* __restrict__ gp) {
    int idx = blockIdx.x * 256 + threadIdx.x;
    if (idx >= BB * C2 * MM) return;
    int m = idx & (MM - 1);
    int c = (idx >> 10) & (C2 - 1);
    int b = idx >> 17;
    int h2 = m >> 5, w2 = m & 31;
    int nb = (h2 * 2) * 64 + w2 * 2;
    size_t base = (size_t)b * NN * C2 + c;
    float a0 = bf2f(gf[base + (size_t)(nb)      * C2]);
    float a1 = bf2f(gf[base + (size_t)(nb + 1)  * C2]);
    float a2 = bf2f(gf[base + (size_t)(nb + 64) * C2]);
    float a3 = bf2f(gf[base + (size_t)(nb + 65) * C2]);
    gp[(size_t)(b * C2 + c) * MM + m] = f2bf(fmaxf(fmaxf(a0, a1), fmaxf(a2, a3)));
}

// ---------- fused attention: softmax(theta^T phi) @ g^T, flash-style ----------
// theta [B][N][32] bf16, phi_p [B][M][32] bf16, g_p [B][128][M] bf16
// -> ag [B][N][128] bf16.  128 threads = 4 waves; block = 16 query rows.
__global__ void attn_kernel(const unsigned short* __restrict__ theta,
                            const unsigned short* __restrict__ phi,
                            const unsigned short* __restrict__ g,
                            unsigned short* __restrict__ ag) {
    __shared__ __align__(16) float sS[16][68];
    __shared__ __align__(16) unsigned short sP[16][64];
    __shared__ float rowM[16], rowS[16], rowScale[16];

    int b  = blockIdx.y;
    int n0 = blockIdx.x * 16;
    int tid = threadIdx.x;
    int w = tid >> 5, lane = tid & 31;
    int r = lane & 15, half = lane >> 4;

    // preload A (theta) fragment: row r, K contiguous: lane half h holds k = 16h..16h+15 (B uses the same order)
    BF16x16 aT;
    {
        const unsigned short* tp = theta + (size_t)(b * NN + n0 + r) * C8 + half * 16;
        aT.q[0] = *(const uint4*)(tp);
        aT.q[1] = *(const uint4*)(tp + 8);
    }
    v8f o0 = {}, o1 = {};
    if (tid < 16) { rowM[tid] = -3.0e38f; rowS[tid] = 0.f; }
    __syncthreads();

    for (int m0 = 0; m0 < MM; m0 += 64) {
        // --- scores chunk: wave w computes cols m0+16w .. +15, K=32 -> one WMMA
        BF16x16 bP;
        {
            int mcol = m0 + w * 16 + r;
            const unsigned short* pp = phi + (size_t)(b * MM + mcol) * C8 + half * 16;
            bP.q[0] = *(const uint4*)(pp);
            bP.q[1] = *(const uint4*)(pp + 8);
        }
        v8f s = {};
        s = wmma_bf16(aT, bP, s);
#pragma unroll
        for (int v = 0; v < 8; ++v) sS[v + half * 8][w * 16 + r] = s[v];
        __syncthreads();

        // --- online softmax update (one thread per query row)
        if (tid < 16) {
            float mx = rowM[tid];
#pragma unroll 4
            for (int j = 0; j < 64; ++j) mx = fmaxf(mx, sS[tid][j]);
            float sc = __expf(rowM[tid] - mx);
            float sum = 0.f;
#pragma unroll 4
            for (int j = 0; j < 64; ++j) {
                float e = __expf(sS[tid][j] - mx);
                sum += e;
                sP[tid][j] = f2bf(e);
            }
            rowS[tid]     = rowS[tid] * sc + sum;
            rowScale[tid] = sc;
            rowM[tid]     = mx;
        }
        __syncthreads();

        // --- rescale running accumulators
#pragma unroll
        for (int v = 0; v < 8; ++v) {
            float sc = rowScale[v + half * 8];
            o0[v] *= sc;
            o1[v] *= sc;
        }

        // --- O += P(16x64) @ G^T chunk : 2 col-tiles x 2 K-steps per wave
#pragma unroll
        for (int kk = 0; kk < 64; kk += 32) {
            BF16x16 aP;
            aP.q[0] = *(const uint4*)(&sP[r][kk + half * 16]);
            aP.q[1] = *(const uint4*)(&sP[r][kk + half * 16 + 8]);
            {
                int c = (w * 2 + 0) * 16 + r;
                const unsigned short* gq = g + (size_t)(b * C2 + c) * MM + m0 + kk + half * 16;
                BF16x16 bG;
                bG.q[0] = *(const uint4*)(gq);
                bG.q[1] = *(const uint4*)(gq + 8);
                o0 = wmma_bf16(aP, bG, o0);
            }
            {
                int c = (w * 2 + 1) * 16 + r;
                const unsigned short* gq = g + (size_t)(b * C2 + c) * MM + m0 + kk + half * 16;
                BF16x16 bG;
                bG.q[0] = *(const uint4*)(gq);
                bG.q[1] = *(const uint4*)(gq + 8);
                o1 = wmma_bf16(aP, bG, o1);
            }
        }
        __syncthreads();
    }

    // --- normalize and store ag [B][N][128]
#pragma unroll
    for (int v = 0; v < 8; ++v) {
        int M = v + half * 8;
        float inv = 1.f / rowS[M];
        size_t rowbase = (size_t)(b * NN + n0 + M) * C2;
        ag[rowbase + (w * 2 + 0) * 16 + r] = f2bf(o0[v] * inv);
        ag[rowbase + (w * 2 + 1) * 16 + r] = f2bf(o1[v] * inv);
    }
}

// ---------- final conv (128->256) + residual: out = x + sigma*(W ag + bias) ----------
__global__ void outconv_kernel(const unsigned short* __restrict__ ag,
                               const unsigned short* __restrict__ wS,
                               const float* __restrict__ bias,
                               const float* __restrict__ x,
                               const float* __restrict__ sigma,
                               float* __restrict__ out) {
    int wid  = blockIdx.x * 8 + (threadIdx.x >> 5);
    int lane = threadIdx.x & 31;
    int r = lane & 15, half = lane >> 4;
    int ot = wid & 15;
    int nt = (wid >> 4) & 255;
    int b  = wid >> 12;
    int n0 = nt * 16;
    int c0 = ot * 16;

    const unsigned short* ap = ag + (size_t)(b * NN + n0 + r) * C2;
    const unsigned short* wb = wS + (size_t)(c0 + r) * C2 + half * 16;
    __builtin_prefetch(wb, 0, 0);

    v8f acc = {};
#pragma unroll
    for (int k0 = 0; k0 < C2; k0 += 32) {
        BF16x16 a;
        a.q[0] = *(const uint4*)(ap + k0 + half * 16);
        a.q[1] = *(const uint4*)(ap + k0 + half * 16 + 8);
        BF16x16 bm;
        bm.q[0] = *(const uint4*)(wb + k0);
        bm.q[1] = *(const uint4*)(wb + k0 + 8);
        acc = wmma_bf16(a, bm, acc);
    }
    float sg = sigma[0];
    float bv = bias[c0 + r];
    size_t base = (size_t)(b * CC + c0 + r) * NN + n0 + half * 8;
    const float4* xp = (const float4*)(x + base);
    float4* op = (float4*)(out + base);
    float4 x0 = xp[0], x1 = xp[1];
    float4 y0, y1;
    y0.x = x0.x + sg * (acc[0] + bv);
    y0.y = x0.y + sg * (acc[1] + bv);
    y0.z = x0.z + sg * (acc[2] + bv);
    y0.w = x0.w + sg * (acc[3] + bv);
    y1.x = x1.x + sg * (acc[4] + bv);
    y1.y = x1.y + sg * (acc[5] + bv);
    y1.z = x1.z + sg * (acc[6] + bv);
    y1.w = x1.w + sg * (acc[7] + bv);
    op[0] = y0;
    op[1] = y1;
}

// ---------- host launch ----------
extern "C" void kernel_launch(void* const* d_in, const int* in_sizes, int n_in,
                              void* d_out, int out_size, void* d_ws, size_t ws_size,
                              hipStream_t stream) {
    const float* x      = (const float*)d_in[0];
    const float* sigma  = (const float*)d_in[1];
    const float* thw    = (const float*)d_in[2];
    const float* thb    = (const float*)d_in[3];
    const float* thu    = (const float*)d_in[4];
    const float* phw    = (const float*)d_in[5];
    const float* phb    = (const float*)d_in[6];
    const float* phu    = (const float*)d_in[7];
    const float* gw     = (const float*)d_in[8];
    const float* gb     = (const float*)d_in[9];
    const float* gu     = (const float*)d_in[10];
    const float* aw     = (const float*)d_in[11];
    const float* ab     = (const float*)d_in[12];
    const float* au     = (const float*)d_in[13];
    float* out          = (float*)d_out;

    char* ws = (char*)d_ws;
    unsigned short* wtheta  = (unsigned short*)(ws + 0);         // 32*256 bf16
    unsigned short* wphi    = (unsigned short*)(ws + 16384);     // 32*256
    unsigned short* wg      = (unsigned short*)(ws + 32768);     // 128*256
    unsigned short* wattn   = (unsigned short*)(ws + 98304);     // 256*128
    unsigned short* xt      = (unsigned short*)(ws + 163840);    // [B][N][256] bf16, 32 MB
    unsigned short* theta   = (unsigned short*)(ws + 33718272);  // [B][N][32]    4 MB
    unsigned short* phifull = (unsigned short*)(ws + 37912576);  // [B][N][32]    4 MB
    unsigned short* gfull   = (unsigned short*)(ws + 42106880);  // [B][N][128]  16 MB
    unsigned short* phip    = (unsigned short*)(ws + 58884096);  // [B][M][32]    1 MB
    unsigned short* gp      = (unsigned short*)(ws + 59932672);  // [B][128][M]   4 MB
    unsigned short* agbuf   = gfull;                             // reuse (dead after pool_g)

    // 1) spectral-norm scaled bf16 weights
    sn_scale_kernel<<<1, 256, 0, stream>>>(thw, thu, wtheta, C8, CC);
    sn_scale_kernel<<<1, 256, 0, stream>>>(phw, phu, wphi,   C8, CC);
    sn_scale_kernel<<<1, 256, 0, stream>>>(gw,  gu,  wg,     C2, CC);
    sn_scale_kernel<<<1, 256, 0, stream>>>(aw,  au,  wattn,  CC, C2);

    // 2) convert+transpose x -> bf16 [B][N][C]
    cvt_transpose_kernel<<<BB * 16 * 64, 256, 0, stream>>>(x, xt);

    // 3) 1x1 convs (WMMA GEMMs)
    conv1x1_kernel<<<(BB * 256 * (C8 / 16)) / 8, 256, 0, stream>>>(xt, wtheta, thb, theta,   C8);
    conv1x1_kernel<<<(BB * 256 * (C8 / 16)) / 8, 256, 0, stream>>>(xt, wphi,   phb, phifull, C8);
    conv1x1_kernel<<<(BB * 256 * (C2 / 16)) / 8, 256, 0, stream>>>(xt, wg,     gb,  gfull,   C2);

    // 4) 2x2 maxpools
    pool_phi_kernel<<<(BB * MM * C8) / 256, 256, 0, stream>>>(phifull, phip);
    pool_g_kernel<<<(BB * C2 * MM) / 256, 256, 0, stream>>>(gfull, gp);

    // 5) fused flash-style attention (writes agbuf; gfull is dead by now)
    {
        dim3 grid(NN / 16, BB);
        attn_kernel<<<grid, 128, 0, stream>>>(theta, phip, gp, agbuf);
    }

    // 6) final conv + residual
    outconv_kernel<<<(BB * 256 * (CC / 16)) / 8, 256, 0, stream>>>(agbuf, wattn, ab, x, sigma, out);
}
